// GraphStackedMultiHeadAttention_22299470201326
// MI455X (gfx1250) — matrix operand order repaired and checked by reading, not compiled, not run on hardware
//
#include <hip/hip_runtime.h>
#include <hip/hip_fp16.h>

typedef __attribute__((ext_vector_type(16))) _Float16 v16h;
typedef __attribute__((ext_vector_type(8)))  _Float16 v8h;
typedef __attribute__((ext_vector_type(8)))  float    v8f;

#define T_    32
#define A_    8
#define N_    128
#define E_    512
#define G_    (T_*A_)       // 256 graphs
#define M_    (G_*N_)       // 32768 nodes
#define ET_   (G_*E_)       // 131072 edges
#define HID_  128
#define KD_   64
#define HEADS_ 4
#define NKT_  (HEADS_*KD_)  // 256 (concat head proj width)

union U16h { v16h v; v8h h[2]; };

// ---------------------------------------------------------------------------
// WMMA GEMM:  C[M x N](f16) = act( A[M x K](f16, optional row gather)
//                                  * Bt(packed col-major [N][K], f16)
//                                  + bias + (accum ? C : 0) )
// Block = 8 waves = 128-row x 64-col macro-tile. The 64-col B panel is staged
// once into LDS (swizzled, conflict-free) and shared by all 8 waves; each wave
// computes a 16x64 tile with v_wmma_f32_16x16x32_f16, A fragments double-
// buffered from global so the next A load overlaps the current WMMA group.
// ---------------------------------------------------------------------------
__global__ void gemm16(const _Float16* __restrict__ A,
                       const int*      __restrict__ rowIdx,
                       const _Float16* __restrict__ Bt,
                       const float*    __restrict__ bias,
                       _Float16*       __restrict__ C,
                       int K, int N, int relu, int accum, int tilesN)
{
    extern __shared__ _Float16 bsh[];          // 64 * K halfs, chunk-swizzled
    const int tid  = threadIdx.x;
    const int rb   = blockIdx.x / tilesN;      // 128-row block
    const int tN   = blockIdx.x % tilesN;      // 64-col block
    const int kch  = K >> 3;                   // 8-half chunks per column
    const int mask = kch - 1;                  // K/8 is a power of two (4/16/64)

    // ---- stage B panel: global -> LDS (ds_store_b128), swizzled by column ----
    const _Float16* bsrc = Bt + (size_t)(tN * 64) * K;
    const int totalChunks = 8 * K;             // 64*K/8
    for (int idx = tid; idx < totalChunks; idx += blockDim.x) {
        int c  = idx / kch;
        int kk = idx - c * kch;
        int dk = (kk + c) & mask;              // rotate chunks within column
        *(v8h*)(bsh + (size_t)c * K + dk * 8) =
            *(const v8h*)(bsrc + (size_t)c * K + kk * 8);
    }
    __syncthreads();

    const int wv   = tid >> 5;
    const int lane = tid & 31;
    const int r    = lane & 15;                // A row within tile / C column
    const int hi   = lane >> 4;                // lane half selects K sub-range

    const int rowBase = rb * 128 + wv * 16;
    int arow = rowBase + r;
    if (rowIdx) arow = rowIdx[arow];           // fused gather (h[send]/h[recv])
    const _Float16* ap = A + (size_t)arow * K;

    const int crow  = rowBase + hi * 8;        // C rows (lane>>4)*8 + j
    const int ccol0 = tN * 64 + r;             // C col  = lane&15 within 16-tile

    v8f acc[4];
    if (accum) {
        #pragma unroll
        for (int t = 0; t < 4; ++t)
            #pragma unroll
            for (int j = 0; j < 8; ++j)
                acc[t][j] = (float)C[(size_t)(crow + j) * N + ccol0 + t * 16];
    } else {
        #pragma unroll
        for (int t = 0; t < 4; ++t)
            #pragma unroll
            for (int j = 0; j < 8; ++j)
                acc[t][j] = 0.0f;
    }

    // A fragment (16x32 f16): lane half hi=0 holds K {0..7,16..23},
    // hi=1 holds K {8..15,24..31} of row r -> two contiguous 16B loads.
    U16h a_cur, a_nxt;
    a_cur.h[0] = *(const v8h*)(ap + hi * 8);
    a_cur.h[1] = *(const v8h*)(ap + 16 + hi * 8);
    a_nxt = a_cur;

    for (int k0 = 0; k0 < K; k0 += 32) {
        if (k0 + 32 < K) {                     // double-buffer next A chunk
            a_nxt.h[0] = *(const v8h*)(ap + k0 + 32 + hi * 8);
            a_nxt.h[1] = *(const v8h*)(ap + k0 + 48 + hi * 8);
        }
        const int ch0 = (k0 >> 3) + hi;        // first 8-half chunk of K range
        #pragma unroll
        for (int t = 0; t < 4; ++t) {
            // B fragment (32x16 f16): lane holds column (lane&15) of the tile,
            // K-half hi, 16 K-values as two swizzled 16B LDS loads.
            const int c = t * 16 + r;
            const _Float16* bp = bsh + (size_t)c * K;
            U16h b;
            b.h[0] = *(const v8h*)(bp + ((ch0 + c) & mask) * 8);
            b.h[1] = *(const v8h*)(bp + ((ch0 + 2 + c) & mask) * 8);
            acc[t] = __builtin_amdgcn_wmma_f32_16x16x32_f16(
                        false, a_cur.v, false, b.v, (short)0, acc[t], false, false);
        }
        a_cur = a_nxt;
    }

    #pragma unroll
    for (int t = 0; t < 4; ++t) {
        int col = ccol0 + t * 16;
        float bv = bias ? bias[col] : 0.0f;
        #pragma unroll
        for (int j = 0; j < 8; ++j) {
            float v = acc[t][j] + bv;
            if (relu) v = v > 0.0f ? v : 0.0f;
            C[(size_t)(crow + j) * N + col] = (_Float16)v;
        }
    }
}

// ---------------------------------------------------------------------------
// Elementwise / prep kernels
// ---------------------------------------------------------------------------
__device__ __forceinline__ unsigned fenc(float f) {
    unsigned u = __float_as_uint(f);
    return (u & 0x80000000u) ? ~u : (u | 0x80000000u);
}
__device__ __forceinline__ float fdec(unsigned u) {
    unsigned v = (u & 0x80000000u) ? (u & 0x7fffffffu) : ~u;
    return __uint_as_float(v);
}

__global__ void k_build_idx(const int* __restrict__ recv, const int* __restrict__ send,
                            int* __restrict__ recvg, int* __restrict__ sendg, int n)
{
    int i = blockIdx.x * blockDim.x + threadIdx.x;
    if (i >= n) return;
    int g = i / E_;
    recvg[i] = recv[i] + g * N_;
    sendg[i] = send[i] + g * N_;
}

__global__ void k_build_x0(const float* __restrict__ nodes, const float* __restrict__ embed,
                           _Float16* __restrict__ x0, int n)  // n = M_*32
{
    int i = blockIdx.x * blockDim.x + threadIdx.x;
    if (i >= n) return;
    int m = i >> 5, c = i & 31;
    float v = 0.0f;
    if (c < 15)      v = nodes[m * 16 + c];
    else if (c < 31) { int ent = (int)nodes[m * 16 + 15]; v = embed[ent * 16 + (c - 15)]; }
    x0[(size_t)m * 32 + c] = (_Float16)v;
}

__global__ void k_build_efin(const float* __restrict__ edges, _Float16* __restrict__ ef, int n)
{   // n = ET_*32, pad 8 -> 32
    int i = blockIdx.x * blockDim.x + threadIdx.x;
    if (i >= n) return;
    int e = i >> 5, c = i & 31;
    ef[i] = (_Float16)(c < 8 ? edges[e * 8 + c] : 0.0f);
}

__global__ void k_pack_dense(const float* __restrict__ W, _Float16* __restrict__ out,
                             int K, int N, int Kpad, int n)  // W[K][N] -> out[N][Kpad]
{
    int i = blockIdx.x * blockDim.x + threadIdx.x;
    if (i >= n) return;
    int c = i / Kpad, k = i % Kpad;
    out[i] = (k < K) ? (_Float16)W[(size_t)k * N + c] : (_Float16)0.0f;
}

__global__ void k_pack_kproj(const float* __restrict__ W, _Float16* __restrict__ out, int n)
{   // W[4][128][64] -> out[256][128], col c = h*64+kk
    int i = blockIdx.x * blockDim.x + threadIdx.x;
    if (i >= n) return;
    int c = i / HID_, d = i % HID_;
    int h = c >> 6, kk = c & 63;
    out[i] = (_Float16)W[((size_t)h * HID_ + d) * KD_ + kk];
}

__global__ void k_bias_add(const float* __restrict__ a, const float* __restrict__ b,
                           float* __restrict__ o, int n)
{
    int i = blockIdx.x * blockDim.x + threadIdx.x;
    if (i < n) o[i] = a[i] + b[i];
}

__global__ void k_init_seg(unsigned* __restrict__ segmax, float* __restrict__ den, int n)
{
    int i = blockIdx.x * blockDim.x + threadIdx.x;
    if (i < n) { segmax[i] = 0u; den[i] = 0.0f; }
}

__global__ void k_zero_f32(float* __restrict__ p, int n)
{
    int i = blockIdx.x * blockDim.x + threadIdx.x;
    if (i < n) p[i] = 0.0f;
}

__global__ void k_logits(const _Float16* __restrict__ ks, const _Float16* __restrict__ kr,
                         const int* __restrict__ recvg, float* __restrict__ logits,
                         unsigned* __restrict__ segmax, int n)  // n = ET_*HEADS_
{
    int i = blockIdx.x * blockDim.x + threadIdx.x;
    if (i >= n) return;
    int e = i >> 2, h = i & 3;
    const _Float16* a = ks + (size_t)e * NKT_ + h * KD_;
    const _Float16* b = kr + (size_t)e * NKT_ + h * KD_;
    float s = 0.0f;
    #pragma unroll
    for (int k = 0; k < KD_; ++k) s += (float)a[k] * (float)b[k];
    s *= 0.125f;                                  // 1/sqrt(64)
    logits[i] = s;
    atomicMax(segmax + recvg[e] * 4 + h, fenc(s));
}

__global__ void k_wex(const float* __restrict__ logits, const unsigned* __restrict__ segmax,
                      const int* __restrict__ recvg, float* __restrict__ wex,
                      float* __restrict__ den, int n)
{
    int i = blockIdx.x * blockDim.x + threadIdx.x;
    if (i >= n) return;
    int e = i >> 2, h = i & 3;
    float m  = fdec(segmax[recvg[e] * 4 + h]);
    float ex = __expf(logits[i] - m);
    wex[i] = ex;
    atomicAdd(den + recvg[e] * 4 + h, ex);
}

__global__ void k_scatter(const _Float16* __restrict__ hh, const int* __restrict__ sendg,
                          const int* __restrict__ recvg, const float* __restrict__ wex,
                          const float* __restrict__ den, float* __restrict__ agg, int n)
{   // n = ET_*HID_ ; per (edge, d): 4 head atomics
    int i = blockIdx.x * blockDim.x + threadIdx.x;
    if (i >= n) return;
    int e = i >> 7, d = i & 127;
    float sv = (float)hh[(size_t)sendg[e] * HID_ + d];
    int rb = recvg[e] * 4;
    size_t ab = (size_t)recvg[e] * (HEADS_ * HID_) + d;
    #pragma unroll
    for (int h = 0; h < HEADS_; ++h) {
        float w = wex[e * 4 + h] / den[rb + h];
        atomicAdd(agg + ab + h * HID_, w * sv);   // global_atomic_add_f32
    }
}

__global__ void k_xnext(const float* __restrict__ agg, _Float16* __restrict__ x, int n)
{
    int i = blockIdx.x * blockDim.x + threadIdx.x;
    if (i < n) x[i] = (_Float16)agg[i];
}

__global__ void k_out(const float* __restrict__ agg, float* __restrict__ out, int n)
{   // n = M_*HID_ ; mean over heads
    int i = blockIdx.x * blockDim.x + threadIdx.x;
    if (i >= n) return;
    int m = i >> 7, d = i & 127;
    const float* a = agg + (size_t)m * (HEADS_ * HID_) + d;
    out[i] = 0.25f * (a[0] + a[HID_] + a[2 * HID_] + a[3 * HID_]);
}

// ---------------------------------------------------------------------------
// Host-side orchestration
// ---------------------------------------------------------------------------
static inline int cdiv(int a, int b) { return (a + b - 1) / b; }

extern "C" void kernel_launch(void* const* d_in, const int* in_sizes, int n_in,
                              void* d_out, int out_size, void* d_ws, size_t ws_size,
                              hipStream_t stream)
{
    (void)in_sizes; (void)n_in; (void)out_size; (void)ws_size;

    const float* nodes   = (const float*)d_in[0];  // [M_,16]
    const float* edges   = (const float*)d_in[1];  // [ET_,8]
    const int*   recvIn  = (const int*)  d_in[2];
    const int*   sendIn  = (const int*)  d_in[3];
    const float* embed   = (const float*)d_in[6];  // [2,16]

    // params: jax pytree leaf order — dict keys sorted per layer:
    // edge_mlp(W1,b1,W2,b2), k_edge(W,b), k_recv(W,b), k_sent(W,b), node_mlp(W1,b1,W2,b2)
    struct LayerP {
        const float *eW1, *eb1, *eW2, *eb2;
        const float *We, *be, *Wr, *br, *Ws, *bs;
        const float *nW1, *nb1, *nW2, *nb2;
    } lp[2];
    for (int li = 0; li < 2; ++li) {
        int b = 7 + li * 14;
        lp[li].eW1 = (const float*)d_in[b + 0];  lp[li].eb1 = (const float*)d_in[b + 1];
        lp[li].eW2 = (const float*)d_in[b + 2];  lp[li].eb2 = (const float*)d_in[b + 3];
        lp[li].We  = (const float*)d_in[b + 4];  lp[li].be  = (const float*)d_in[b + 5];
        lp[li].Wr  = (const float*)d_in[b + 6];  lp[li].br  = (const float*)d_in[b + 7];
        lp[li].Ws  = (const float*)d_in[b + 8];  lp[li].bs  = (const float*)d_in[b + 9];
        lp[li].nW1 = (const float*)d_in[b + 10]; lp[li].nb1 = (const float*)d_in[b + 11];
        lp[li].nW2 = (const float*)d_in[b + 12]; lp[li].nb2 = (const float*)d_in[b + 13];
    }
    const int nKin[2] = {32, HEADS_ * HID_};   // padded node-mlp input widths

    // ---- workspace arena (lifetime-aliased) ----
    char* p = (char*)d_ws;
    auto alloc = [&](size_t bytes) { char* r = p; p += (bytes + 255) & ~(size_t)255; return r; };

    _Float16* wp = (_Float16*)alloc(2u * 1024 * 1024);     // packed weights
    size_t wo = 0;
    auto wal = [&](size_t nelem) { _Float16* r = wp + wo; wo += nelem; return r; };
    _Float16 *nW1t[2], *nW2t[2], *eW1t[2], *eW2t[2], *Wst[2], *Wrt[2], *Wet[2];
    for (int li = 0; li < 2; ++li) {
        nW1t[li] = wal((size_t)HID_ * nKin[li]);
        nW2t[li] = wal((size_t)HID_ * HID_);
        eW1t[li] = wal((size_t)HID_ * 32);
        eW2t[li] = wal((size_t)HID_ * HID_);
        Wst[li]  = wal((size_t)NKT_ * HID_);
        Wrt[li]  = wal((size_t)NKT_ * HID_);
        Wet[li]  = wal((size_t)NKT_ * HID_);
    }
    float* cb[2];
    cb[0] = (float*)alloc(NKT_ * sizeof(float));
    cb[1] = (float*)alloc(NKT_ * sizeof(float));

    int* sendg = (int*)alloc((size_t)ET_ * 4);
    int* recvg = (int*)alloc((size_t)ET_ * 4);
    _Float16* efin = (_Float16*)alloc((size_t)ET_ * 32 * 2);       // persistent
    _Float16* t1   = (_Float16*)alloc((size_t)ET_ * HID_ * 2);     // mlp hidden (node+edge)
    _Float16* hh   = (_Float16*)alloc((size_t)M_ * HID_ * 2);      // node mlp output
    _Float16* efh  = (_Float16*)alloc((size_t)ET_ * HID_ * 2);     // edge mlp output
    char* regKS = alloc((size_t)ET_ * NKT_ * 2);   // == M_*512*4 : ks_h, later agg(f32)
    char* regKR = alloc((size_t)ET_ * NKT_ * 2);   // kr_h, also x0/x1 activations
    _Float16* ksh = (_Float16*)regKS;
    _Float16* krh = (_Float16*)regKR;
    float*    agg = (float*)   regKS;              // alias: ks dead after logits
    _Float16* xh  = (_Float16*)regKR;              // alias: kr dead after logits
    float*    logits = (float*)alloc((size_t)ET_ * HEADS_ * 4);
    float*    wex    = (float*)alloc((size_t)ET_ * HEADS_ * 4);
    unsigned* segmax = (unsigned*)alloc((size_t)M_ * HEADS_ * 4);
    float*    den    = (float*)alloc((size_t)M_ * HEADS_ * 4);

    const int TB = 256;
    auto gemm = [&](const _Float16* A, const int* idx, const _Float16* Bt, const float* bias,
                    _Float16* C, int Mrows, int K, int N, int relu, int accum) {
        int tilesN = N / 64;
        int grid   = (Mrows / 128) * tilesN;           // block = 128 rows x 64 cols
        size_t shb = (size_t)64 * K * sizeof(_Float16); // LDS B panel
        gemm16<<<grid, TB, shb, stream>>>(A, idx, Bt, bias, C, K, N, relu, accum, tilesN);
    };

    // ---- weight packing ----
    for (int li = 0; li < 2; ++li) {
        int kin = (li == 0) ? 31 : HEADS_ * HID_;
        k_pack_dense<<<cdiv(HID_ * nKin[li], TB), TB, 0, stream>>>(lp[li].nW1, nW1t[li], kin, HID_, nKin[li], HID_ * nKin[li]);
        k_pack_dense<<<cdiv(HID_ * HID_, TB), TB, 0, stream>>>(lp[li].nW2, nW2t[li], HID_, HID_, HID_, HID_ * HID_);
        k_pack_dense<<<cdiv(HID_ * 32, TB), TB, 0, stream>>>(lp[li].eW1, eW1t[li], 8, HID_, 32, HID_ * 32);
        k_pack_dense<<<cdiv(HID_ * HID_, TB), TB, 0, stream>>>(lp[li].eW2, eW2t[li], HID_, HID_, HID_, HID_ * HID_);
        k_pack_kproj<<<cdiv(NKT_ * HID_, TB), TB, 0, stream>>>(lp[li].Ws, Wst[li], NKT_ * HID_);
        k_pack_kproj<<<cdiv(NKT_ * HID_, TB), TB, 0, stream>>>(lp[li].Wr, Wrt[li], NKT_ * HID_);
        k_pack_kproj<<<cdiv(NKT_ * HID_, TB), TB, 0, stream>>>(lp[li].We, Wet[li], NKT_ * HID_);
        k_bias_add<<<1, NKT_, 0, stream>>>(lp[li].br, lp[li].be, cb[li], NKT_);
    }

    // ---- graph-global indices + padded inputs ----
    k_build_idx<<<cdiv(ET_, TB), TB, 0, stream>>>(recvIn, sendIn, recvg, sendg, ET_);
    k_build_x0<<<cdiv(M_ * 32, TB), TB, 0, stream>>>(nodes, embed, xh, M_ * 32);
    k_build_efin<<<cdiv(ET_ * 32, TB), TB, 0, stream>>>(edges, efin, ET_ * 32);

    // ---- layers ----
    for (int li = 0; li < 2; ++li) {
        // node MLP: x -> t1 -> hh
        gemm(xh, nullptr, nW1t[li], lp[li].nb1, t1, M_, nKin[li], HID_, 1, 0);
        gemm(t1, nullptr, nW2t[li], lp[li].nb2, hh, M_, HID_, HID_, 1, 0);
        // edge MLP: efin -> t1 -> efh
        gemm(efin, nullptr, eW1t[li], lp[li].eb1, t1, ET_, 32, HID_, 1, 0);
        gemm(t1, nullptr, eW2t[li], lp[li].eb2, efh, ET_, HID_, HID_, 1, 0);
        // k projections (gather fused via rowIdx)
        gemm(hh, sendg, Wst[li], lp[li].bs, ksh, ET_, HID_, NKT_, 0, 0);        // ks
        gemm(hh, recvg, Wrt[li], cb[li], krh, ET_, HID_, NKT_, 0, 0);           // kr  = ra*Wr + br+be
        gemm(efh, nullptr, Wet[li], nullptr, krh, ET_, HID_, NKT_, 0, 1);       // kr += ef*We
        // segment softmax
        k_init_seg<<<cdiv(M_ * HEADS_, TB), TB, 0, stream>>>(segmax, den, M_ * HEADS_);
        k_logits<<<cdiv(ET_ * HEADS_, TB), TB, 0, stream>>>(ksh, krh, recvg, logits, segmax, ET_ * HEADS_);
        k_wex<<<cdiv(ET_ * HEADS_, TB), TB, 0, stream>>>(logits, segmax, recvg, wex, den, ET_ * HEADS_);
        // message aggregation (agg aliases ks region — ks dead after logits)
        k_zero_f32<<<cdiv(M_ * HEADS_ * HID_, TB), TB, 0, stream>>>(agg, M_ * HEADS_ * HID_);
        k_scatter<<<cdiv(ET_ * HID_, TB), TB, 0, stream>>>(hh, sendg, recvg, wex, den, agg, ET_ * HID_);
        if (li == 0)
            k_xnext<<<cdiv(M_ * HEADS_ * HID_, TB), TB, 0, stream>>>(agg, xh, M_ * HEADS_ * HID_);
        else
            k_out<<<cdiv(M_ * HID_, TB), TB, 0, stream>>>(agg, (float*)d_out, M_ * HID_);
    }
}
